// ThorMoE_57518202028602
// MI455X (gfx1250) — compile-verified, matrix operand order checked by reading, not run on hardware
//
#include <hip/hip_runtime.h>
#include <hip/hip_bf16.h>

// ---------------------------------------------------------------------------
// Grouped MoE MLP: y[e] = (x[e] @ W1[e] + b1[e]) @ W2[e] + b2[e]
// E=16, T=512, H=1024, I=4096.  fp32 in / fp32 out, bf16 WMMA compute with
// fp32 accumulation (v_wmma_f32_16x16x32_bf16), intermediate h kept as bf16
// in d_ws.  Double-buffered LDS (one barrier/K-step); GEMM2's bf16 A operand
// is moved global->LDS with CDNA5 async loads (ASYNCcnt).
// ---------------------------------------------------------------------------

typedef __attribute__((ext_vector_type(16))) __bf16 v16bf;
typedef __attribute__((ext_vector_type(8)))  __bf16 v8bf;
typedef __attribute__((ext_vector_type(4)))  __bf16 v4bf;
typedef __attribute__((ext_vector_type(8)))  float  v8f;
typedef __attribute__((ext_vector_type(4)))  float  v4f;

// The async-to-LDS builtin takes pointers to 16-byte int vectors:
//   param0: v4i __device__(AS1)*   param1: v4i AS(3)*
typedef int v4i_vs __attribute__((vector_size(16)));
typedef __attribute__((address_space(1))) v4i_vs* gv4i_p;
typedef __attribute__((address_space(3))) v4i_vs* lv4i_p;

constexpr int BM = 128;       // block tile M (tokens)
constexpr int BN = 128;       // block tile N
constexpr int KB = 32;        // K per step == WMMA K for bf16
constexpr int LDSK = 40;      // padded LDS row stride (halfwords): 80 B, 16B-aligned
constexpr int NTHREADS = 256; // 8 wave32

#if defined(__has_builtin)
#if __has_builtin(__builtin_amdgcn_global_load_async_to_lds_b128)
#define HAVE_ASYNC_B128 1
#endif
#if __has_builtin(__builtin_amdgcn_s_wait_asynccnt)
#define HAVE_WAIT_ASYNC 1
#endif
#endif

// Async global->LDS, 16 bytes per lane.  AS casts go through uintptr_t
// (plain int->ptr casts); for LDS the low 32 bits of the flat address are
// the LDS byte offset.
__device__ inline void async_copy_b128(const __bf16* g, __bf16* l) {
#if defined(HAVE_ASYNC_B128)
    __builtin_amdgcn_global_load_async_to_lds_b128(
        (gv4i_p)(uintptr_t)g,
        (lv4i_p)(uintptr_t)(unsigned)(uintptr_t)l,
        /*offset=*/0, /*cpol=*/0);
#else
    unsigned loff = (unsigned)(uintptr_t)l;
    asm volatile("global_load_async_to_lds_b128 %0, %1, off"
                 :: "v"(loff), "v"(g) : "memory");
#endif
}

__device__ inline void wait_async0() {
#if defined(HAVE_WAIT_ASYNC)
    __builtin_amdgcn_s_wait_asynccnt(0);
#else
    asm volatile("s_wait_asynccnt 0x0" ::: "memory");
#endif
}

// 16x32 bf16 WMMA operand fragment from an LDS tile (row-major [r][k]).
// ISA layout (CDNA5 ISA): lanes 0-15 -> row=lane, K 0..7 & 16..23;
// lanes 16-31 -> row=lane-16, K 8..15 & 24..31  => two 16-B chunks.
__device__ inline v16bf load_frag(const __bf16* p) {
    v8bf lo = *(const v8bf*)(p);
    v8bf hi = *(const v8bf*)(p + 16);
    v16bf f;
#pragma unroll
    for (int i = 0; i < 8; ++i) { f[i] = lo[i]; f[i + 8] = hi[i]; }
    return f;
}

template <bool ABF16, bool CBF16>
__global__ __launch_bounds__(NTHREADS)
void moe_bgemm_bf16wmma(const void* __restrict__ Ag,   // [E][M][K] f32 or bf16
                        const float* __restrict__ Bg,  // [E][K][N] f32
                        const float* __restrict__ bias,// [E][N]    f32
                        void* __restrict__ Cg,         // [E][M][N] bf16 or f32
                        int M, int N, int K)
{
    __shared__ __align__(16) __bf16 As[2][BM][LDSK];  // [m][k]
    __shared__ __align__(16) __bf16 Bs[2][BN][LDSK];  // [n][k] (transposed)

    const int e   = blockIdx.z;
    const int m0  = blockIdx.y * BM;
    const int n0  = blockIdx.x * BN;    // N fast-varying -> A tile shared in L2
    const int tid = threadIdx.x;
    const int lane = tid & 31;
    const int wave = tid >> 5;
    const int wm   = wave & 3;          // 4 waves along M (32 rows each)
    const int wn   = wave >> 2;         // 2 waves along N (64 cols each)
    const int lrow = lane & 15;
    const int kb   = (lane >> 4) * 8;   // 0 or 8 (halfwords)

    const size_t eA = (size_t)e * M * K;
    const float* Bexp = Bg + (size_t)e * K * N;
    const int ksteps = K / KB;

    // staging registers (B always; A only for the f32->bf16 path)
    v4f a_f32[4];
    v4f b_f32[4];

    auto load_global = [&](int kt, int buf) {
        if constexpr (ABF16) {
            // bf16 A: direct async global->LDS, 2x b128 per thread
            const __bf16* Aexp = (const __bf16*)Ag + eA;
#pragma unroll
            for (int s = 0; s < 2; ++s) {
                int idx = tid + s * NTHREADS;          // 0..511 chunks
                int row = idx >> 2;                    // 4 x 16B chunks per row
                int c8  = idx & 3;
                const __bf16* g = Aexp + (size_t)(m0 + row) * K + kt * KB + c8 * 8;
                async_copy_b128(g, &As[buf][row][c8 * 8]);
            }
        } else {
#pragma unroll
            for (int s = 0; s < 4; ++s) {
                int idx = tid + s * NTHREADS;          // 0..1023
                int row = idx >> 3;                    // 8 float4 per row
                int c4  = idx & 7;
                const float* A = (const float*)Ag + eA + (size_t)(m0 + row) * K
                               + kt * KB + c4 * 4;
                a_f32[s] = *(const v4f*)A;
            }
        }
        // B tile: KB x BN floats = 32 rows x 32 float4 -> regs
#pragma unroll
        for (int s = 0; s < 4; ++s) {
            int idx = tid + s * NTHREADS;
            int kr  = idx >> 5;
            int c4  = idx & 31;
            b_f32[s] = *(const v4f*)(Bexp + (size_t)(kt * KB + kr) * N + n0 + c4 * 4);
        }
        if (kt + 2 < ksteps)
            __builtin_prefetch(Bexp + (size_t)((kt + 2) * KB + (tid >> 5)) * N
                               + n0 + (tid & 31) * 4, 0, 1);
    };

    auto stash_lds = [&](int buf) {
        if constexpr (!ABF16) {
#pragma unroll
            for (int s = 0; s < 4; ++s) {
                int idx = tid + s * NTHREADS;
                int row = idx >> 3;
                int c4  = idx & 7;
                v4f f = a_f32[s];
                v4bf v;
                v[0] = (__bf16)f[0]; v[1] = (__bf16)f[1];
                v[2] = (__bf16)f[2]; v[3] = (__bf16)f[3];
                *(v4bf*)&As[buf][row][c4 * 4] = v;
            }
        }
#pragma unroll
        for (int s = 0; s < 4; ++s) {   // B: convert + transpose into [n][k]
            int idx = tid + s * NTHREADS;
            int kr  = idx >> 5;
            int c4  = idx & 31;
            v4f f = b_f32[s];
#pragma unroll
            for (int j = 0; j < 4; ++j)
                Bs[buf][c4 * 4 + j][kr] = (__bf16)f[j];
        }
    };

    // accumulators: 2 (M) x 4 (N) 16x16 tiles per wave
    v8f acc[2][4];
#pragma unroll
    for (int mi = 0; mi < 2; ++mi)
#pragma unroll
        for (int ni = 0; ni < 4; ++ni)
            acc[mi][ni] = (v8f)0.0f;

    // ---- double-buffered pipelined main loop ------------------------------
    load_global(0, 0);
    stash_lds(0);
    if constexpr (ABF16) wait_async0();
    __syncthreads();

    for (int kt = 0; kt < ksteps; ++kt) {
        const int cur = kt & 1;
        const int nxt = cur ^ 1;
        const bool havenext = (kt + 1 < ksteps);

        if (havenext) load_global(kt + 1, nxt);   // async A + B regs

        v16bf afr[2], bfr[4];
#pragma unroll
        for (int mi = 0; mi < 2; ++mi)
            afr[mi] = load_frag(&As[cur][wm * 32 + mi * 16 + lrow][kb]);
#pragma unroll
        for (int ni = 0; ni < 4; ++ni)
            bfr[ni] = load_frag(&Bs[cur][wn * 64 + ni * 16 + lrow][kb]);

#pragma unroll
        for (int mi = 0; mi < 2; ++mi)
#pragma unroll
            for (int ni = 0; ni < 4; ++ni)
                acc[mi][ni] = __builtin_amdgcn_wmma_f32_16x16x32_bf16(
                    false, afr[mi], false, bfr[ni],
                    (short)0, acc[mi][ni], false, false);

        if (havenext) {
            stash_lds(nxt);
            if constexpr (ABF16) wait_async0();
        }
        __syncthreads();
    }

    // ---- epilogue: bias add + store ---------------------------------------
    // C layout (32-bit 16x16): lanes 0-15: M=vgpr, N=lane; lanes 16-31: M=8+vgpr.
    const int lhi = lane >> 4;
#pragma unroll
    for (int mi = 0; mi < 2; ++mi) {
#pragma unroll
        for (int ni = 0; ni < 4; ++ni) {
            const int gr0 = m0 + wm * 32 + mi * 16 + lhi * 8;
            const int gc  = n0 + wn * 64 + ni * 16 + lrow;
            const float bv = bias[(size_t)e * N + gc];
#pragma unroll
            for (int r = 0; r < 8; ++r) {
                float v = acc[mi][ni][r] + bv;
                size_t off = (size_t)e * M * N + (size_t)(gr0 + r) * N + gc;
                if constexpr (CBF16)
                    ((__bf16*)Cg)[off] = (__bf16)v;
                else
                    ((float*)Cg)[off] = v;
            }
        }
    }
}

extern "C" void kernel_launch(void* const* d_in, const int* in_sizes, int n_in,
                              void* d_out, int out_size, void* d_ws, size_t ws_size,
                              hipStream_t stream) {
    constexpr int E = 16, T = 512, H = 1024, I = 4096;

    const float* x  = (const float*)d_in[0];  // [E][T][H]
    const float* W1 = (const float*)d_in[1];  // [E][H][I]
    const float* b1 = (const float*)d_in[2];  // [E][I]
    const float* W2 = (const float*)d_in[3];  // [E][I][H]
    const float* b2 = (const float*)d_in[4];  // [E][H]

    __bf16* h = (__bf16*)d_ws;                // [E][T][I] bf16, 64 MiB scratch

    // GEMM1: h = x @ W1 + b1   (M=T, N=I, K=H), A f32, C bf16
    dim3 g1(I / BN, T / BM, E);
    moe_bgemm_bf16wmma<false, true><<<g1, NTHREADS, 0, stream>>>(
        (const void*)x, W1, b1, (void*)h, T, I, H);

    // GEMM2: y = h @ W2 + b2   (M=T, N=H, K=I), A bf16, C f32 -> d_out (E*T, H)
    dim3 g2(H / BN, T / BM, E);
    moe_bgemm_bf16wmma<true, false><<<g2, NTHREADS, 0, stream>>>(
        (const void*)h, W2, b2, d_out, T, H, I);
}
